// Selection_83408264888653
// MI455X (gfx1250) — matrix operand order repaired and checked by reading, not compiled, not run on hardware
//
#include <hip/hip_runtime.h>
#include <math.h>

typedef __attribute__((ext_vector_type(16))) _Float16 v16h;
typedef __attribute__((ext_vector_type(8)))  _Float16 v8h;
typedef __attribute__((ext_vector_type(8)))  float    v8f;
typedef __attribute__((ext_vector_type(4)))  float    v4f;

#define LN_EPS 1e-5f
#define INV_NORM 0.08838834764831845f   /* 1/sqrt(128) */

// ---------------- WMMA fragment loaders (wave32, 16x16x32 f16) ----------------
// A (16x32, row-major f16, leading dim lda):
//   lanes 0-15:  M=lane,    K = kb+[0..15]
//   lanes 16-31: M=lane-16, K = kb+[16..31]
__device__ __forceinline__ v16h load_a_frag(const _Float16* __restrict__ A, int lda,
                                            int row_base, int kb, int lane) {
  const _Float16* p = A + (size_t)(row_base + (lane & 15)) * lda + kb + ((lane >> 4) << 4);
  v8h lo = *(const v8h*)p;
  v8h hi = *(const v8h*)(p + 8);
  v16h r;
  #pragma unroll
  for (int i = 0; i < 8; ++i) { r[i] = lo[i]; r[i + 8] = hi[i]; }
  return r;
}
// B (32x16) loaded from N-major ("Bt[n][k]") f16 with leading dim ldb:
//   lanes 0-15:  N=lane,    K = kb+[0..15]
//   lanes 16-31: N=lane-16, K = kb+[16..31]
__device__ __forceinline__ v16h load_b_frag(const _Float16* __restrict__ Bt, int ldb,
                                            int col_base, int kb, int lane) {
  const _Float16* p = Bt + (size_t)(col_base + (lane & 15)) * ldb + kb + ((lane >> 4) << 4);
  return *(const v16h*)p;
}

#define WMMA_F16(A_, B_, C_) \
  __builtin_amdgcn_wmma_f32_16x16x32_f16(false, (A_), false, (B_), (short)0, (C_), false, false)

// ---------------- Kernel 5 (defined first so disasm window shows it):
// S = weights(4096x512) @ spriteT^T(512x16384), permuted store to (P,B,4,H,W).
// Grid (128,32), block 256 = 8 waves arranged 4(M) x 2(N): block tile 128x128
// (~256 KB A+B per block -> ~1 GB total L2 traffic; B reused 4x inside WGP$).
// Wave tile: 32(M) x 64(N) = 2x4 WMMA subtiles. Ping-pong double-buffered K loop:
// two disjoint fragment buffer sets, prefetch clause for the next K-step issued
// before each WMMA burst -> waits are "previous clause done", never 0.
__global__ void __launch_bounds__(256) k_sgemm(const _Float16* __restrict__ w16,
                                               const _Float16* __restrict__ Bt,
                                               float* __restrict__ S_out) {
  int tid = threadIdx.x, w = tid >> 5, lane = tid & 31;
  int wrow = w >> 1, wcol = w & 1;
  int nBase = blockIdx.x * 128 + wcol * 64;
  int mBase = blockIdx.y * 128 + wrow * 32;
  v8f acc[2][4] = {};

  v16h a0_0, a1_0, b0_0, b1_0, b2_0, b3_0;   // buffer 0
  v16h a0_1, a1_1, b0_1, b1_1, b2_1, b3_1;   // buffer 1

  // prologue: buffer 0 <- kb = 0
  a0_0 = load_a_frag(w16, 512, mBase,      0, lane);
  a1_0 = load_a_frag(w16, 512, mBase + 16, 0, lane);
  b0_0 = load_b_frag(Bt, 512, nBase,      0, lane);
  b1_0 = load_b_frag(Bt, 512, nBase + 16, 0, lane);
  b2_0 = load_b_frag(Bt, 512, nBase + 32, 0, lane);
  b3_0 = load_b_frag(Bt, 512, nBase + 48, 0, lane);

  #pragma unroll 1
  for (int kb = 0; kb < 448; kb += 64) {
    // buffer 1 <- kb+32 (in flight during buffer-0 WMMAs)
    a0_1 = load_a_frag(w16, 512, mBase,      kb + 32, lane);
    a1_1 = load_a_frag(w16, 512, mBase + 16, kb + 32, lane);
    b0_1 = load_b_frag(Bt, 512, nBase,      kb + 32, lane);
    b1_1 = load_b_frag(Bt, 512, nBase + 16, kb + 32, lane);
    b2_1 = load_b_frag(Bt, 512, nBase + 32, kb + 32, lane);
    b3_1 = load_b_frag(Bt, 512, nBase + 48, kb + 32, lane);

    acc[0][0] = WMMA_F16(a0_0, b0_0, acc[0][0]);
    acc[1][0] = WMMA_F16(a1_0, b0_0, acc[1][0]);
    acc[0][1] = WMMA_F16(a0_0, b1_0, acc[0][1]);
    acc[1][1] = WMMA_F16(a1_0, b1_0, acc[1][1]);
    acc[0][2] = WMMA_F16(a0_0, b2_0, acc[0][2]);
    acc[1][2] = WMMA_F16(a1_0, b2_0, acc[1][2]);
    acc[0][3] = WMMA_F16(a0_0, b3_0, acc[0][3]);
    acc[1][3] = WMMA_F16(a1_0, b3_0, acc[1][3]);

    // buffer 0 <- kb+64 (in flight during buffer-1 WMMAs)
    a0_0 = load_a_frag(w16, 512, mBase,      kb + 64, lane);
    a1_0 = load_a_frag(w16, 512, mBase + 16, kb + 64, lane);
    b0_0 = load_b_frag(Bt, 512, nBase,      kb + 64, lane);
    b1_0 = load_b_frag(Bt, 512, nBase + 16, kb + 64, lane);
    b2_0 = load_b_frag(Bt, 512, nBase + 32, kb + 64, lane);
    b3_0 = load_b_frag(Bt, 512, nBase + 48, kb + 64, lane);

    acc[0][0] = WMMA_F16(a0_1, b0_1, acc[0][0]);
    acc[1][0] = WMMA_F16(a1_1, b0_1, acc[1][0]);
    acc[0][1] = WMMA_F16(a0_1, b1_1, acc[0][1]);
    acc[1][1] = WMMA_F16(a1_1, b1_1, acc[1][1]);
    acc[0][2] = WMMA_F16(a0_1, b2_1, acc[0][2]);
    acc[1][2] = WMMA_F16(a1_1, b2_1, acc[1][2]);
    acc[0][3] = WMMA_F16(a0_1, b3_1, acc[0][3]);
    acc[1][3] = WMMA_F16(a1_1, b3_1, acc[1][3]);
  }
  // epilogue: kb = 448 (buffer 0) with prefetch of kb = 480, then kb = 480 (buffer 1)
  a0_1 = load_a_frag(w16, 512, mBase,      480, lane);
  a1_1 = load_a_frag(w16, 512, mBase + 16, 480, lane);
  b0_1 = load_b_frag(Bt, 512, nBase,      480, lane);
  b1_1 = load_b_frag(Bt, 512, nBase + 16, 480, lane);
  b2_1 = load_b_frag(Bt, 512, nBase + 32, 480, lane);
  b3_1 = load_b_frag(Bt, 512, nBase + 48, 480, lane);

  acc[0][0] = WMMA_F16(a0_0, b0_0, acc[0][0]);
  acc[1][0] = WMMA_F16(a1_0, b0_0, acc[1][0]);
  acc[0][1] = WMMA_F16(a0_0, b1_0, acc[0][1]);
  acc[1][1] = WMMA_F16(a1_0, b1_0, acc[1][1]);
  acc[0][2] = WMMA_F16(a0_0, b2_0, acc[0][2]);
  acc[1][2] = WMMA_F16(a1_0, b2_0, acc[1][2]);
  acc[0][3] = WMMA_F16(a0_0, b3_0, acc[0][3]);
  acc[1][3] = WMMA_F16(a1_0, b3_0, acc[1][3]);

  acc[0][0] = WMMA_F16(a0_1, b0_1, acc[0][0]);
  acc[1][0] = WMMA_F16(a1_1, b0_1, acc[1][0]);
  acc[0][1] = WMMA_F16(a0_1, b1_1, acc[0][1]);
  acc[1][1] = WMMA_F16(a1_1, b1_1, acc[1][1]);
  acc[0][2] = WMMA_F16(a0_1, b2_1, acc[0][2]);
  acc[1][2] = WMMA_F16(a1_1, b2_1, acc[1][2]);
  acc[0][3] = WMMA_F16(a0_1, b3_1, acc[0][3]);
  acc[1][3] = WMMA_F16(a1_1, b3_1, acc[1][3]);

  // epilogue: n -> (c, hw); m = b*128+p -> out row p*32+b; write (P,B,4,H,W)
  int c  = nBase >> 12;
  int n0 = nBase & 4095;
  int rbase = (lane >> 4) << 3;
  #pragma unroll
  for (int i = 0; i < 2; ++i) {
    #pragma unroll
    for (int t = 0; t < 4; ++t) {
      int hw = n0 + t * 16 + (lane & 15);
      #pragma unroll
      for (int r = 0; r < 8; ++r) {
        int m = mBase + i * 16 + rbase + r;
        int b = m >> 7, p = m & 127;
        S_out[((size_t)((p * 32 + b) * 4 + c)) * 4096 + hw] = acc[i][t][r];
      }
    }
  }
}

// ---------------- Kernel 1: lat = LN(latents @ linear_w^T + b) -> f16 (512x128, rows 510/511 zero)
__global__ void k_latents(const float* __restrict__ latents, const float* __restrict__ linear_w,
                          const float* __restrict__ linear_b, _Float16* __restrict__ lat16) {
  int k = blockIdx.x, z = threadIdx.x;
  if (k >= 510) { lat16[(size_t)k * 128 + z] = (_Float16)0.f; return; }
  __shared__ float lrow[256];
  __shared__ float red[128];
  lrow[z]       = latents[(size_t)k * 256 + z];
  lrow[z + 128] = latents[(size_t)k * 256 + 128 + z];
  __syncthreads();
  float acc = linear_b[z];
  #pragma unroll 4
  for (int i = 0; i < 256; ++i) acc = fmaf(lrow[i], linear_w[(size_t)z * 256 + i], acc);
  red[z] = acc; __syncthreads();
  for (int s = 64; s > 0; s >>= 1) { if (z < s) red[z] += red[z + s]; __syncthreads(); }
  float mean = red[0] * (1.f / 128.f);
  __syncthreads();
  float d = acc - mean;
  red[z] = d * d; __syncthreads();
  for (int s = 64; s > 0; s >>= 1) { if (z < s) red[z] += red[z + s]; __syncthreads(); }
  float var = red[0] * (1.f / 128.f);
  lat16[(size_t)k * 128 + z] = (_Float16)(d * rsqrtf(var + LN_EPS));
}

// ---------------- Kernel 2: a = LN(xp @ anchors_w^T + b) -> f16 (4096x128); dots[m]
__global__ void k_anchors(const float* __restrict__ x, const float* __restrict__ anchors_w,
                          const float* __restrict__ anchors_b, const float* __restrict__ blank,
                          _Float16* __restrict__ a16, float* __restrict__ dots) {
  int m = blockIdx.x, z = threadIdx.x;
  int b = m >> 7, p = m & 127;
  __shared__ float xrow[128];
  __shared__ float red[128];
  xrow[z] = x[((size_t)(b * 128 + z)) * 128 + p];
  __syncthreads();
  float acc = anchors_b[z];
  #pragma unroll 4
  for (int c = 0; c < 128; ++c) acc = fmaf(xrow[c], anchors_w[(size_t)z * 128 + c], acc);
  red[z] = acc; __syncthreads();
  for (int s = 64; s > 0; s >>= 1) { if (z < s) red[z] += red[z + s]; __syncthreads(); }
  float mean = red[0] * (1.f / 128.f);
  __syncthreads();
  float d = acc - mean;
  red[z] = d * d; __syncthreads();
  for (int s = 64; s > 0; s >>= 1) { if (z < s) red[z] += red[z + s]; __syncthreads(); }
  float var = red[0] * (1.f / 128.f);
  a16[(size_t)m * 128 + z] = (_Float16)(d * rsqrtf(var + LN_EPS));
  __syncthreads();
  red[z] = xrow[z] * blank[(p & 1) * 128 + z];
  __syncthreads();
  for (int s = 64; s > 0; s >>= 1) { if (z < s) red[z] += red[z + s]; __syncthreads(); }
  if (z == 0) dots[m] = red[0];
}

// ---------------- Kernel 3: spriteT[n][k] f16 (16384x512); fuses concat+zero rows+f16+transpose
__global__ void k_spriteT(const float* __restrict__ prot, const float* __restrict__ masks,
                          _Float16* __restrict__ Bt) {
  int nBase = blockIdx.x * 256;   // never crosses a channel boundary (256 | 4096)
  int kBase = blockIdx.y * 32;
  int tid = threadIdx.x;
  __shared__ _Float16 ldsT[256][40];   // 40 f16 = 80B rows (16B aligned, bank spread)
  int kk = tid >> 3;                   // 0..31
  int q  = tid & 7;                    // 8 threads per k-row, 32 n each
  int k = kBase + kk;
  int c = nBase >> 12;
  int hw0 = (nBase & 4095) + q * 32;
  const float* src = nullptr;
  if (k < 510)
    src = (c < 3) ? (prot + ((size_t)k * 3 + c) * 4096 + hw0)
                  : (masks + (size_t)k * 4096 + hw0);
  #pragma unroll
  for (int j = 0; j < 8; ++j) {
    v4f v = {0.f, 0.f, 0.f, 0.f};
    if (src) v = *(const v4f*)(src + j * 4);
    int n0 = q * 32 + j * 4;
    ldsT[n0 + 0][kk] = (_Float16)v.x;
    ldsT[n0 + 1][kk] = (_Float16)v.y;
    ldsT[n0 + 2][kk] = (_Float16)v.z;
    ldsT[n0 + 3][kk] = (_Float16)v.w;
  }
  __syncthreads();
  _Float16* dst = Bt + (size_t)(nBase + tid) * 512 + kBase;
  const v8h* row = (const v8h*)(&ldsT[tid][0]);
  #pragma unroll
  for (int i = 0; i < 4; ++i) ((v8h*)dst)[i] = row[i];
}

// ---------------- Kernel 4: WMMA logits GEMM + blank patch + softmax/log-softmax fusion
// Block: 256 thr = 8 waves; 16 rows x 512 cols. Wave w covers cols [w*64, w*64+64).
// Ping-pong double-buffered over the 4 K-steps (fully unrolled).
__global__ void __launch_bounds__(256) k_logits(
    const _Float16* __restrict__ a16, const _Float16* __restrict__ lat16,
    const float* __restrict__ dots,
    float* __restrict__ out_weights, float* __restrict__ out_logits,
    float* __restrict__ out_logp, _Float16* __restrict__ w16) {
  int mTile = blockIdx.x * 16;
  int tid = threadIdx.x;
  int w = tid >> 5, lane = tid & 31;
  __shared__ float ls[16][512];
  __shared__ float red[256];
  __shared__ float rowmax[16], rowsum[16];

  v8f acc[4] = {};
  v16h abuf[2];
  v16h bbuf[2][4];
  abuf[0] = load_a_frag(a16, 128, mTile, 0, lane);
  #pragma unroll
  for (int t = 0; t < 4; ++t) bbuf[0][t] = load_b_frag(lat16, 128, w * 64 + t * 16, 0, lane);
  #pragma unroll
  for (int s = 0; s < 4; ++s) {
    int cur = s & 1, nxt = cur ^ 1;
    if (s < 3) {
      abuf[nxt] = load_a_frag(a16, 128, mTile, (s + 1) * 32, lane);
      #pragma unroll
      for (int t = 0; t < 4; ++t)
        bbuf[nxt][t] = load_b_frag(lat16, 128, w * 64 + t * 16, (s + 1) * 32, lane);
    }
    #pragma unroll
    for (int t = 0; t < 4; ++t)
      acc[t] = WMMA_F16(abuf[cur], bbuf[cur][t], acc[t]);
  }
  // C/D layout: VGPR r -> M = r + (lane>=16 ? 8 : 0), N = lane%16
  int rbase = (lane >> 4) << 3;
  #pragma unroll
  for (int t = 0; t < 4; ++t) {
    int col = w * 64 + t * 16 + (lane & 15);
    #pragma unroll
    for (int r = 0; r < 8; ++r) ls[rbase + r][col] = acc[t][r] * INV_NORM;
  }
  __syncthreads();
  if (tid < 16) {   // blank logit columns (510,511): dots / NORM, duplicated
    float v = dots[mTile + tid] * INV_NORM;
    ls[tid][510] = v; ls[tid][511] = v;
  }
  __syncthreads();
  // row-wise softmax: 16 threads per row, 32 cols each
  int row = tid >> 4, c16 = tid & 15;
  float mx = -1e30f;
  for (int j = 0; j < 32; ++j) mx = fmaxf(mx, ls[row][c16 + j * 16]);
  red[tid] = mx; __syncthreads();
  if (c16 == 0) { float m2 = red[tid]; for (int t = 1; t < 16; ++t) m2 = fmaxf(m2, red[tid + t]); rowmax[row] = m2; }
  __syncthreads();
  mx = rowmax[row];
  float s = 0.f;
  for (int j = 0; j < 32; ++j) s += __expf(ls[row][c16 + j * 16] - mx);
  red[tid] = s; __syncthreads();
  if (c16 == 0) { float s2 = 0.f; for (int t = 0; t < 16; ++t) s2 += red[tid + t]; rowsum[row] = s2; }
  __syncthreads();
  s = rowsum[row];
  float inv_s = 1.f / s, lse = mx + __logf(s);
  int m = mTile + row, b = m >> 7, p = m & 127;
  size_t mrow = (size_t)m * 512;
  size_t lprow = (size_t)(p * 32 + b) * 512;
  for (int j = 0; j < 32; ++j) {
    int col = c16 + j * 16;
    float v = ls[row][col];
    float wgt = __expf(v - mx) * inv_s;
    out_logits[mrow + col]  = v;
    out_weights[mrow + col] = wgt;
    w16[mrow + col]         = (_Float16)wgt;
    out_logp[lprow + col]   = v - lse;
  }
}

extern "C" void kernel_launch(void* const* d_in, const int* in_sizes, int n_in,
                              void* d_out, int out_size, void* d_ws, size_t ws_size,
                              hipStream_t stream) {
  (void)in_sizes; (void)n_in; (void)out_size; (void)ws_size;
  const float* x         = (const float*)d_in[0];
  const float* latents   = (const float*)d_in[1];
  const float* blank     = (const float*)d_in[2];
  const float* linear_w  = (const float*)d_in[3];
  const float* linear_b  = (const float*)d_in[4];
  const float* anchors_w = (const float*)d_in[5];
  const float* anchors_b = (const float*)d_in[6];
  const float* prot      = (const float*)d_in[7];
  const float* masks     = (const float*)d_in[8];

  float* out         = (float*)d_out;
  float* out_weights = out;                 // (4096, 512)
  float* out_logits  = out + 2097152;       // (4096, 512)
  float* out_logp    = out + 4194304;       // (128, 32, 512)
  float* out_S       = out + 6291456;       // (128, 32, 4, 64, 64)

  char* ws = (char*)d_ws;                   // total ~22.2 MB
  _Float16* lat16 = (_Float16*)(ws + 0);          // 512*128*2
  _Float16* a16   = (_Float16*)(ws + 131072);     // 4096*128*2
  float*    dots  = (float*)(ws + 1179648);       // 4096*4
  _Float16* w16   = (_Float16*)(ws + 1196032);    // 4096*512*2
  _Float16* Bt    = (_Float16*)(ws + 5390336);    // 16384*512*2

  k_latents<<<512, 128, 0, stream>>>(latents, linear_w, linear_b, lat16);
  k_anchors<<<4096, 128, 0, stream>>>(x, anchors_w, anchors_b, blank, a16, dots);
  k_spriteT<<<dim3(64, 16), 256, 0, stream>>>(prot, masks, Bt);
  k_logits<<<256, 256, 0, stream>>>(a16, lat16, dots, out_weights, out_logits, out_logp, w16);
  k_sgemm<<<dim3(128, 32), 256, 0, stream>>>(w16, Bt, out_S);
}
